// RSSM_89309549953056
// MI455X (gfx1250) — matrix operand order repaired and checked by reading, not compiled, not run on hardware
//
#include <hip/hip_runtime.h>

// ---------------- problem constants ----------------
#define Bb   256
#define Tt   64
#define Hh   512
#define Dd   2048            // 4*H
#define Ss   32
#define Cc   32
#define Aa   32
#define Ee   1024
#define Kk   8
#define BSs  256             // D/K
#define H3   1536            // 3*H
#define SC   1024            // S*C
#define XIN  1056            // S*C + A
#define XINP 1088            // XIN padded to multiple of 64
#define DE   3072            // D + E
#define G3   768             // 3*BS
#define GIW  6144            // K*3*BS

typedef __bf16 v16bf __attribute__((ext_vector_type(16)));
typedef float  v8f   __attribute__((ext_vector_type(8)));

union Frag  { v16bf v; uint4 u[2]; };
union Acc8  { v8f   v; float f[8]; };

__device__ __forceinline__ unsigned short f2bf(float f) {
  union { __bf16 b; unsigned short u; } cv; cv.b = (__bf16)f; return cv.u;
}
__device__ __forceinline__ float fsigmoid(float x) {
  return __builtin_amdgcn_rcpf(1.0f + __expf(-x));   // v_exp_f32 + v_rcp_f32
}
__device__ __forceinline__ float fsilu(float x) { return x * fsigmoid(x); }
__device__ __forceinline__ float ftanh(float x) { return 2.0f * fsigmoid(2.0f * x) - 1.0f; }

// CDNA5 async global->LDS copy (GVS mode: SGPR64 base + per-lane u32 byte offset).
// Tracked by ASYNCcnt; drain with s_wait_asynccnt before the barrier.
__device__ __forceinline__ void async_b128(unsigned lds_off, unsigned goff,
                                           const void* base) {
  asm volatile("global_load_async_to_lds_b128 %0, %1, %2"
               :: "v"(lds_off), "v"(goff), "s"(base) : "memory");
}
__device__ __forceinline__ void wait_async0() {
  asm volatile("s_wait_asynccnt 0x0" ::: "memory");
}

// ---------------- fp32 -> bf16 weight conversion ----------------
__global__ void cvt_bf16_k(const float* __restrict__ src,
                           unsigned short* __restrict__ dst, long n) {
  long i = (long)blockIdx.x * blockDim.x + threadIdx.x;
  long stride = (long)gridDim.x * blockDim.x;
  for (; i < n; i += stride) dst[i] = f2bf(src[i]);
}

// fp32 [rows,K] -> bf16 [rows,Kpad] with zero pad
__global__ void cvt_pad_k(const float* __restrict__ src,
                          unsigned short* __restrict__ dst,
                          long rows, int Kin, int Kpad) {
  long i = (long)blockIdx.x * blockDim.x + threadIdx.x;
  long stride = (long)gridDim.x * blockDim.x;
  long n = rows * (long)Kpad;
  for (; i < n; i += stride) {
    long r = i / Kpad;
    int  c = (int)(i - r * Kpad);
    dst[i] = (c < Kin) ? f2bf(src[r * Kin + c]) : (unsigned short)0;
  }
}

// ---------------- state init ----------------
__global__ void init_k(const float* __restrict__ d0, const float* __restrict__ s0,
                       float* __restrict__ deter_f, unsigned short* __restrict__ deter_bf,
                       float* __restrict__ stoch) {
  int idx = blockIdx.x * blockDim.x + threadIdx.x;
  if (idx < Bb * Dd) { float v = d0[idx]; deter_f[idx] = v; deter_bf[idx] = f2bf(v); }
  if (idx < Bb * SC) stoch[idx] = s0[idx];
}

// ---------------- per-step prep: mask state, build bf16 [stoch|action|pad] ----------------
__global__ void prep_k(float* __restrict__ deter_f, unsigned short* __restrict__ deter_bf,
                       float* __restrict__ stoch,
                       const float* __restrict__ actions, const int* __restrict__ is_first,
                       int t, unsigned short* __restrict__ xin_bf) {
  int idx = blockIdx.x * blockDim.x + threadIdx.x;
  if (idx >= Bb * Dd) return;
  int b = idx >> 11;          // / 2048
  int j = idx & 2047;
  bool first = is_first[b * Tt + t] > 0;
  float dv = first ? 0.0f : deter_f[idx];
  deter_f[idx]  = dv;
  deter_bf[idx] = f2bf(dv);
  if (j < SC) {
    float s = first ? 0.0f : stoch[b * SC + j];
    stoch[b * SC + j] = s;
    xin_bf[b * XINP + j] = f2bf(s);
  } else if (j < XIN) {
    xin_bf[b * XINP + j] = f2bf(actions[(long)b * Tt * Aa + (long)t * Aa + (j - SC)]);
  } else if (j < XINP) {
    xin_bf[b * XINP + j] = 0;   // K padding
  }
}

// ---------------- WMMA GEMM: act(A[M,K]bf16 * W[N,K]bf16^T + bias) ----------------
// 128 threads / 4 waves; block tile 128x128; wave tile 64x64 (4x4 frags);
// K staged 64-wide through LDS via async global->LDS b128; 32 v_wmma per stage.
// Outputs: optional fp32 Cf and/or bf16 Cb (same ldc).
__global__ __launch_bounds__(128) void gemm_bf16_k(
    const unsigned short* __restrict__ A, int lda,
    const unsigned short* __restrict__ W, int ldw,
    const float* __restrict__ bias,
    float* __restrict__ Cf, unsigned short* __restrict__ Cb, long ldc,
    int Kdim, int act,
    long zA, long zW, long zB, long zC) {
  __shared__ __align__(16) unsigned short sA[128][72];
  __shared__ __align__(16) unsigned short sB[128][72];

  A    += (long)blockIdx.z * zA;
  W    += (long)blockIdx.z * zW;
  bias += (long)blockIdx.z * zB;
  if (Cf) Cf += (long)blockIdx.z * zC;
  if (Cb) Cb += (long)blockIdx.z * zC;

  const int m0   = blockIdx.y * 128;
  const int n0   = blockIdx.x * 128;
  const int tid  = threadIdx.x;
  const int lane = tid & 31;
  const int wid  = tid >> 5;
  const int wm   = (wid >> 1) * 64;      // wave row origin in tile
  const int wn   = (wid & 1) * 64;       // wave col origin in tile

  const int am = lane & 15;              // row/col within 16
  const int ah = lane >> 4;              // lane half

  Acc8 acc[4][4];
  #pragma unroll
  for (int i = 0; i < 4; ++i)
    #pragma unroll
    for (int j = 0; j < 4; ++j)
      #pragma unroll
      for (int e = 0; e < 8; ++e) acc[i][j].f[e] = 0.0f;

  for (int k0 = 0; k0 < Kdim; k0 += 64) {
    // --- async stage A: 128 rows x 64 bf16 (8 chunks of 16B per row) ---
    #pragma unroll
    for (int i = 0; i < 8; ++i) {
      int c    = tid + 128 * i;
      int row  = c >> 3;
      int colc = (c & 7) * 8;            // bf16 elements
      unsigned gof = (unsigned)(((long)(m0 + row) * lda + k0 + colc) * 2);
      unsigned lof = (unsigned)(size_t)&sA[row][colc];
      async_b128(lof, gof, A);
    }
    // --- async stage B: 128 rows x 64 bf16 ---
    #pragma unroll
    for (int i = 0; i < 8; ++i) {
      int c    = tid + 128 * i;
      int row  = c >> 3;
      int colc = (c & 7) * 8;
      unsigned gof = (unsigned)(((long)(n0 + row) * ldw + k0 + colc) * 2);
      unsigned lof = (unsigned)(size_t)&sB[row][colc];
      async_b128(lof, gof, W);
    }
    wait_async0();
    __syncthreads();

    // --- two 32-wide WMMA substeps over the 64-wide stage ---
    #pragma unroll
    for (int ks = 0; ks < 64; ks += 32) {
      Frag a[4], b[4];
      #pragma unroll
      for (int i = 0; i < 4; ++i) {
        a[i].u[0] = *(const uint4*)&sA[wm + 16 * i + am][ks + ah * 8];
        a[i].u[1] = *(const uint4*)&sA[wm + 16 * i + am][ks + 16 + ah * 8];
      }
      #pragma unroll
      for (int j = 0; j < 4; ++j) {
        b[j].u[0] = *(const uint4*)&sB[wn + 16 * j + am][ks + ah * 8];
        b[j].u[1] = *(const uint4*)&sB[wn + 16 * j + am][ks + 16 + ah * 8];
      }
      #pragma unroll
      for (int i = 0; i < 4; ++i)
        #pragma unroll
        for (int j = 0; j < 4; ++j)
          acc[i][j].v = __builtin_amdgcn_wmma_f32_16x16x32_bf16(
              false, a[i].v, false, b[j].v, (short)0, acc[i][j].v, false, false);
    }
    __syncthreads();
  }

  // --- epilogue: bias + optional SiLU, fp32 and/or bf16 stores ---
  #pragma unroll
  for (int i = 0; i < 4; ++i) {
    #pragma unroll
    for (int j = 0; j < 4; ++j) {
      int col  = n0 + wn + 16 * j + am;
      float bv = bias[col];
      #pragma unroll
      for (int e = 0; e < 8; ++e) {
        int row = m0 + wm + 16 * i + e + ah * 8;
        float v = acc[i][j].f[e] + bv;
        if (act) v = fsilu(v);
        long o = (long)row * ldc + col;
        if (Cf) Cf[o] = v;
        if (Cb) Cb[o] = f2bf(v);
      }
    }
  }
}

// ---------------- GRU gate fusion ----------------
__global__ void gru_k(const float* __restrict__ gi, const float* __restrict__ gh,
                      const float* __restrict__ hblk,
                      float* __restrict__ deter_f, unsigned short* __restrict__ deter_bf,
                      float* __restrict__ out_deter, long out_stride) {
  int idx = blockIdx.x * blockDim.x + threadIdx.x;
  if (idx >= Bb * Dd) return;
  int b = idx >> 11;
  int r = idx & 2047;
  int k = r >> 8;
  int c = r & 255;
  long gb = (long)b * GIW + (long)k * G3;
  float ir = gi[gb + c], iz = gi[gb + BSs + c], in_ = gi[gb + 2 * BSs + c];
  float hr = gh[gb + c], hz = gh[gb + BSs + c], hn  = gh[gb + 2 * BSs + c];
  float h  = hblk[idx];
  float rr = fsigmoid(ir + hr);
  float z  = fsigmoid(iz + hz);
  float n  = ftanh(in_ + rr * hn);
  float dn = (1.0f - z) * n + z * h;
  deter_f[idx]  = dn;
  deter_bf[idx] = f2bf(dn);
  out_deter[(long)b * out_stride + r] = dn;
}

// ---------------- concat [deter | embed_t] -> bf16 ----------------
__global__ void concat_k(const float* __restrict__ deter_f, const float* __restrict__ embed,
                         int t, unsigned short* __restrict__ dein_bf) {
  int idx = blockIdx.x * blockDim.x + threadIdx.x;
  if (idx >= Bb * DE) return;
  int b = idx / DE;
  int j = idx - b * DE;
  float v = (j < Dd) ? deter_f[b * Dd + j]
                     : embed[(long)b * Tt * Ee + (long)t * Ee + (j - Dd)];
  dein_bf[idx] = f2bf(v);
}

// ---------------- softmax + unimix + categorical sample -> one-hot ----------------
__global__ void sample_k(const float* __restrict__ post, long ldpost,
                         float* __restrict__ stoch, float* __restrict__ out_stoch,
                         long ldout, int t) {
  int idx = blockIdx.x * blockDim.x + threadIdx.x;
  if (idx >= Bb * Ss) return;
  int b = idx >> 5;
  int s = idx & 31;
  const float* lg = post + (long)b * ldpost + s * Cc;
  float mx = -1e30f;
  #pragma unroll
  for (int c = 0; c < Cc; ++c) mx = fmaxf(mx, lg[c]);
  float p[Cc]; float sum = 0.0f;
  #pragma unroll
  for (int c = 0; c < Cc; ++c) { p[c] = __expf(lg[c] - mx); sum += p[c]; }
  float inv = __builtin_amdgcn_rcpf(sum);
  unsigned h = (unsigned)t * 1315423911u ^ (unsigned)b * 2654435761u ^ (unsigned)s * 97531u;
  h ^= h >> 16; h *= 0x7feb352du; h ^= h >> 15; h *= 0x846ca68bu; h ^= h >> 16;
  float u = (float)(h >> 8) * (1.0f / 16777216.0f);
  float cum = 0.0f; int pick = Cc - 1;
  #pragma unroll
  for (int c = 0; c < Cc; ++c) {
    float pr = 0.99f * p[c] * inv + 0.01f / (float)Cc;
    cum += pr;
    if (u < cum) { pick = c; break; }
  }
  long so = (long)b * SC + s * Cc;
  long oo = (long)b * ldout + s * Cc;
  #pragma unroll
  for (int c = 0; c < Cc; ++c) {
    float o = (c == pick) ? 1.0f : 0.0f;
    stoch[so + c] = o;
    out_stoch[oo + c] = o;
  }
}

// ---------------- host ----------------
extern "C" void kernel_launch(void* const* d_in, const int* in_sizes, int n_in,
                              void* d_out, int out_size, void* d_ws, size_t ws_size,
                              hipStream_t stream) {
  const float* embed    = (const float*)d_in[0];
  const float* actions  = (const float*)d_in[1];
  const int*   is_first = (const int*)  d_in[2];
  const float* deter0   = (const float*)d_in[3];
  const float* stoch0   = (const float*)d_in[4];
  const float* W_in   = (const float*)d_in[5];  const float* b_in   = (const float*)d_in[6];
  const float* W_blk  = (const float*)d_in[7];  const float* b_blk  = (const float*)d_in[8];
  const float* W_ih   = (const float*)d_in[9];  const float* b_ih   = (const float*)d_in[10];
  const float* W_hh   = (const float*)d_in[11]; const float* b_hh   = (const float*)d_in[12];
  const float* W1     = (const float*)d_in[13]; const float* b1     = (const float*)d_in[14];
  const float* W2     = (const float*)d_in[15]; const float* b2     = (const float*)d_in[16];
  const float* W3     = (const float*)d_in[17]; const float* b3     = (const float*)d_in[18];
  const float* W_proj = (const float*)d_in[19]; const float* b_proj = (const float*)d_in[20];

  float* out       = (float*)d_out;
  float* out_deter = out;
  float* out_stoch = out_deter + (size_t)Bb * Tt * Dd;
  float* out_prior = out_stoch + (size_t)Bb * Tt * SC;
  float* out_post  = out_prior + (size_t)Bb * Tt * SC;

  char* base = (char*)d_ws;
  size_t off = 0;
  auto alloc = [&](size_t bytes) -> char* {
    char* p = base + off;
    off = (off + bytes + 255) & ~(size_t)255;
    return p;
  };
  unsigned short* Win_bf   = (unsigned short*)alloc((size_t)H3 * XINP * 2);
  unsigned short* Wblk_bf  = (unsigned short*)alloc((size_t)Dd * Dd * 2);
  unsigned short* Wih_bf   = (unsigned short*)alloc((size_t)GIW * H3 * 2);
  unsigned short* Whh_bf   = (unsigned short*)alloc((size_t)GIW * BSs * 2);
  unsigned short* W1_bf    = (unsigned short*)alloc((size_t)Hh * Dd * 2);
  unsigned short* W2_bf    = (unsigned short*)alloc((size_t)Hh * Hh * 2);
  unsigned short* W3_bf    = (unsigned short*)alloc((size_t)SC * Hh * 2);
  unsigned short* Wproj_bf = (unsigned short*)alloc((size_t)Dd * DE * 2);
  float*          deter_f  = (float*)         alloc((size_t)Bb * Dd * 4);
  unsigned short* deter_bf = (unsigned short*)alloc((size_t)Bb * Dd * 2);
  float*          stoch    = (float*)         alloc((size_t)Bb * SC * 4);
  unsigned short* xin_bf   = (unsigned short*)alloc((size_t)Bb * XINP * 2);
  unsigned short* x_bf     = (unsigned short*)alloc((size_t)Bb * H3 * 2);
  float*          hblk_f   = (float*)         alloc((size_t)Bb * Dd * 4);
  unsigned short* hblk_bf  = (unsigned short*)alloc((size_t)Bb * Dd * 2);  // reused as de
  float*          gi       = (float*)         alloc((size_t)Bb * GIW * 4);
  float*          gh       = (float*)         alloc((size_t)Bb * GIW * 4);
  unsigned short* h1_bf    = (unsigned short*)alloc((size_t)Bb * Hh * 2);
  unsigned short* h2_bf    = (unsigned short*)alloc((size_t)Bb * Hh * 2);
  unsigned short* dein_bf  = (unsigned short*)alloc((size_t)Bb * DE * 2);
  (void)ws_size; (void)in_sizes; (void)n_in; (void)out_size;

  auto cvt = [&](const float* s, unsigned short* d, long n) {
    cvt_bf16_k<<<2048, 256, 0, stream>>>(s, d, n);
  };
  cvt_pad_k<<<2048, 256, 0, stream>>>(W_in, Win_bf, (long)H3, XIN, XINP);
  cvt(W_blk,  Wblk_bf,  (long)Dd * Dd);
  cvt(W_ih,   Wih_bf,   (long)GIW * H3);
  cvt(W_hh,   Whh_bf,   (long)GIW * BSs);
  cvt(W1,     W1_bf,    (long)Hh * Dd);
  cvt(W2,     W2_bf,    (long)Hh * Hh);
  cvt(W3,     W3_bf,    (long)SC * Hh);
  cvt(W_proj, Wproj_bf, (long)Dd * DE);

  init_k<<<(Bb * Dd + 255) / 256, 256, 0, stream>>>(deter0, stoch0, deter_f, deter_bf, stoch);

  auto gemm = [&](const unsigned short* Ap, int lda, const unsigned short* Wp, int ldw,
                  const float* bp, float* Cf, unsigned short* Cb, long ldc,
                  int M, int N, int Kd, int act,
                  int Z, long zA, long zW, long zB, long zC) {
    dim3 g(N / 128, M / 128, Z);
    gemm_bf16_k<<<g, 128, 0, stream>>>(Ap, lda, Wp, ldw, bp, Cf, Cb, ldc, Kd, act,
                                       zA, zW, zB, zC);
  };

  const long ldo_sc = (long)Tt * SC;
  const long ldo_d  = (long)Tt * Dd;

  for (int t = 0; t < Tt; ++t) {
    prep_k<<<(Bb * Dd + 255) / 256, 256, 0, stream>>>(deter_f, deter_bf, stoch,
                                                      actions, is_first, t, xin_bf);

    // x = silu(xin @ W_in^T + b_in)            -> bf16 only
    gemm(xin_bf, XINP, Win_bf, XINP, b_in, nullptr, x_bf, H3, Bb, H3, XINP, 1,
         1, 0, 0, 0, 0);
    // hblk = deter @ W_blk^T + b_blk           -> f32 (GRU) + bf16 (W_hh input)
    gemm(deter_bf, Dd, Wblk_bf, Dd, b_blk, hblk_f, hblk_bf, Dd, Bb, Dd, Dd, 0,
         1, 0, 0, 0, 0);
    // gi = x @ W_ih^T + b_ih                   -> f32 only
    gemm(x_bf, H3, Wih_bf, H3, b_ih, gi, nullptr, GIW, Bb, GIW, H3, 0,
         1, 0, 0, 0, 0);
    // gh[k] = hblk[:,k] @ W_hh[k]^T + b_hh[k]  block-diagonal, gridDim.z = 8
    gemm(hblk_bf, Dd, Whh_bf, BSs, b_hh, gh, nullptr, GIW, Bb, G3, BSs, 0,
         Kk, (long)BSs, (long)G3 * BSs, (long)G3, (long)G3);
    gru_k<<<(Bb * Dd + 255) / 256, 256, 0, stream>>>(gi, gh, hblk_f, deter_f, deter_bf,
                                                     out_deter + (long)t * Dd, ldo_d);

    // prior MLP -> out_prior[:, t]
    gemm(deter_bf, Dd, W1_bf, Dd, b1, nullptr, h1_bf, Hh, Bb, Hh, Dd, 1, 1, 0, 0, 0, 0);
    gemm(h1_bf, Hh, W2_bf, Hh, b2, nullptr, h2_bf, Hh, Bb, Hh, Hh, 1, 1, 0, 0, 0, 0);
    gemm(h2_bf, Hh, W3_bf, Hh, b3, out_prior + (long)t * SC, nullptr, ldo_sc,
         Bb, SC, Hh, 0, 1, 0, 0, 0, 0);

    // de = [deter | embed_t] @ W_proj^T + b_proj   (bf16, reuse hblk_bf)
    concat_k<<<(Bb * DE + 255) / 256, 256, 0, stream>>>(deter_f, embed, t, dein_bf);
    gemm(dein_bf, DE, Wproj_bf, DE, b_proj, nullptr, hblk_bf, Dd, Bb, Dd, DE, 0,
         1, 0, 0, 0, 0);

    // post MLP -> out_post[:, t]
    gemm(hblk_bf, Dd, W1_bf, Dd, b1, nullptr, h1_bf, Hh, Bb, Hh, Dd, 1, 1, 0, 0, 0, 0);
    gemm(h1_bf, Hh, W2_bf, Hh, b2, nullptr, h2_bf, Hh, Bb, Hh, Hh, 1, 1, 0, 0, 0, 0);
    gemm(h2_bf, Hh, W3_bf, Hh, b3, out_post + (long)t * SC, nullptr, ldo_sc,
         Bb, SC, Hh, 0, 1, 0, 0, 0, 0);

    sample_k<<<(Bb * Ss + 255) / 256, 256, 0, stream>>>(out_post + (long)t * SC, ldo_sc,
                                                        stoch, out_stoch + (long)t * SC,
                                                        ldo_sc, t);
  }
}
